// GATWildfire_70772471103953
// MI455X (gfx1250) — matrix operand order repaired and checked by reading, not compiled, not run on hardware
//
#include <hip/hip_runtime.h>

// Problem constants (from reference)
#define INF_ 61
#define KP_ 64         // padded K for input projection
#define HD_ 256        // hidden
#define HEADS_ 8
#define CPH_ 32        // channels per head
#define LAYERS_ 4
#define EPS_ 1e-5f

// ---------------- WMMA fragment types (CDNA5 / gfx1250, wave32) -------------
typedef __attribute__((ext_vector_type(16))) __bf16          bf16x16;
typedef __attribute__((ext_vector_type(8)))  float           f32x8;
typedef __attribute__((ext_vector_type(8)))  unsigned short  u16x8;

__device__ __forceinline__ unsigned short f2bf(float f) {
  unsigned u = __float_as_uint(f);
  unsigned r = u + 0x7FFFu + ((u >> 16) & 1u);   // round-to-nearest-even
  return (unsigned short)(r >> 16);
}
// order-preserving float <-> uint encoding for atomic max on floats
__device__ __forceinline__ unsigned encf(float f) {
  unsigned u = __float_as_uint(f);
  return (u & 0x80000000u) ? ~u : (u | 0x80000000u);
}
__device__ __forceinline__ float decf(unsigned e) {
  unsigned u = (e & 0x80000000u) ? (e & 0x7FFFFFFFu) : ~e;
  return __uint_as_float(u);
}
#define ENC_NEG_INF 0x007FFFFFu   // encf(-inf)

// ---------------- utility fills ----------------
__global__ void fill_f32(float* p, float v, long n) {
  long t = (long)blockIdx.x * blockDim.x + threadIdx.x;
  if (t < n) p[t] = v;
}
__global__ void fill_u32(unsigned* p, unsigned v, long n) {
  long t = (long)blockIdx.x * blockDim.x + threadIdx.x;
  if (t < n) p[t] = v;
}

// ---------------- conversions ----------------
// x [N,61] f32 -> xb [N,64] bf16 (zero padded)
__global__ void cvt_x(const float* __restrict__ x, unsigned short* __restrict__ xb, int N) {
  long t = (long)blockIdx.x * blockDim.x + threadIdx.x;
  if (t >= (long)N * KP_) return;
  int n = (int)(t >> 6), k = (int)(t & 63);
  xb[t] = (k < INF_) ? f2bf(x[(long)n * INF_ + k]) : (unsigned short)0;
}
// W [K,NC] row-major f32 -> Wt [NC,KP] bf16 transposed (zero padded in K)
__global__ void cvt_wt(const float* __restrict__ W, unsigned short* __restrict__ Wt,
                       int K, int NC, int KPAD) {
  long t = (long)blockIdx.x * blockDim.x + threadIdx.x;
  if (t >= (long)NC * KPAD) return;
  int n = (int)(t / KPAD), k = (int)(t % KPAD);
  Wt[t] = (k < K) ? f2bf(W[(long)k * NC + n]) : (unsigned short)0;
}
__global__ void cvt_h(const float* __restrict__ h, unsigned short* __restrict__ hb, long cnt) {
  long t = (long)blockIdx.x * blockDim.x + threadIdx.x;
  if (t < cnt) hb[t] = f2bf(h[t]);
}

// ---------------- WMMA GEMM:  C[M,NC] = A[M,K](bf16) * Bt[NC,K](bf16)^T + bias
// One wave per 16x16 output tile; K stepped by 32 via v_wmma_f32_16x16x32_bf16.
__global__ void gemm_bf16(const unsigned short* __restrict__ A,
                          const unsigned short* __restrict__ Bt,
                          const float* __restrict__ bias,
                          float* __restrict__ C, int M, int K, int NC) {
  int lane = threadIdx.x & 31;
  long wave = ((long)blockIdx.x * blockDim.x + threadIdx.x) >> 5;
  int tilesN = NC >> 4;
  long tiles = (long)(M >> 4) * tilesN;
  if (wave >= tiles) return;
  int tm = (int)(wave / tilesN), tn = (int)(wave % tilesN);
  int half = lane >> 4;        // K-half selector per CDNA5 16-bit layouts
  int r16  = lane & 15;
  const unsigned short* arow = A  + (long)(tm * 16 + r16) * K;   // A: row m = lane%16
  const unsigned short* brow = Bt + (long)(tn * 16 + r16) * K;   // B: col n = lane%16
  f32x8 acc = {};
  for (int k0 = 0; k0 < K; k0 += 32) {
    union { u16x8 u[2]; bf16x16 v; } fa, fb;
    // A 16x32 bf16: frag elems 0..7 -> K = k0+half*8 .. +7 ; 8..15 -> +16
    fa.u[0] = *(const u16x8*)(arow + k0 + half * 8);
    fa.u[1] = *(const u16x8*)(arow + k0 + half * 8 + 16);
    // B 32x16 bf16: frag elems 0..7 -> K = k0+half*8 .. +7 ; 8..15 -> +16
    fb.u[0] = *(const u16x8*)(brow + k0 + half * 8);
    fb.u[1] = *(const u16x8*)(brow + k0 + half * 8 + 16);
    acc = __builtin_amdgcn_wmma_f32_16x16x32_bf16(false, fa.v, false, fb.v,
                                                  (short)0, acc, false, false);
  }
  int n = tn * 16 + r16;
  float bv = bias ? bias[n] : 0.0f;
  int mbase = tm * 16 + half * 8;        // C/D layout: VGPR r -> M = half*8 + r
  #pragma unroll
  for (int r = 0; r < 8; ++r)
    C[(long)(mbase + r) * NC + n] = acc[r] + bv;
}

// ---------------- attention coefficients: per (node, head) dots ----------------
__global__ void attn_coef(const float* __restrict__ z, const float* __restrict__ asrc,
                          const float* __restrict__ adst, float* __restrict__ es,
                          float* __restrict__ ed, int N) {
  int t = blockIdx.x * blockDim.x + threadIdx.x;
  if (t >= N * HEADS_) return;
  int n = t >> 3, hh = t & 7;
  const float* zr = z + (long)n * HD_ + hh * CPH_;
  const float* as = asrc + hh * CPH_;
  const float* ad = adst + hh * CPH_;
  float s = 0.f, d = 0.f;
  #pragma unroll
  for (int c = 0; c < CPH_; ++c) { float v = zr[c]; s += v * as[c]; d += v * ad[c]; }
  es[t] = s; ed[t] = d;
}

// ---------------- edge pass A: logits + segment max ----------------
__global__ void edge_logits(const int* __restrict__ ei, const float* __restrict__ es,
                            const float* __restrict__ ed, float* __restrict__ ee,
                            unsigned* __restrict__ nmax, int E, int N) {
  long t = (long)blockIdx.x * blockDim.x + threadIdx.x;
  long tot = (long)(E + N) * HEADS_;
  if (t >= tot) return;
  int e = (int)(t >> 3), hh = (int)(t & 7);
  int s = (e < E) ? ei[e] : (e - E);           // self loops appended
  int d = (e < E) ? ei[E + e] : (e - E);
  float x = es[s * HEADS_ + hh] + ed[d * HEADS_ + hh];
  x = (x > 0.f) ? x : 0.2f * x;                // leaky_relu(0.2)
  ee[t] = x;
  atomicMax(&nmax[d * HEADS_ + hh], encf(x));
}

// ---------------- edge pass B: softmax numerators + weighted scatter ----------------
// one wave per edge; lane l handles channels [l*8, l*8+8), head = l/4
__global__ void edge_scatter(const int* __restrict__ ei, const float* __restrict__ ee,
                             const unsigned* __restrict__ nmax, const float* __restrict__ z,
                             float* __restrict__ nsum, float* __restrict__ agg,
                             int E, int N) {
  int lane = threadIdx.x & 31;
  long wave = ((long)blockIdx.x * blockDim.x + threadIdx.x) >> 5;
  if (wave >= (long)(E + N)) return;
  int e = (int)wave;
  int s = (e < E) ? ei[e] : (e - E);
  int d = (e < E) ? ei[E + e] : (e - E);
  int hh = lane >> 2;
  float p = __expf(ee[(long)e * HEADS_ + hh] - decf(nmax[d * HEADS_ + hh]));
  if ((lane & 3) == 0) atomicAdd(&nsum[d * HEADS_ + hh], p);
  int c0 = lane * 8;
  const float* zr = z + (long)s * HD_ + c0;
  float* ar = agg + (long)d * HD_ + c0;
  #pragma unroll
  for (int k = 0; k < 8; ++k) atomicAdd(&ar[k], p * zr[k]);
}

// ---------------- finalize aggregation: divide by denom, add conv bias ----------------
__global__ void finalize_agg(const float* __restrict__ agg, const float* __restrict__ nsum,
                             const float* __restrict__ bias, float* __restrict__ pre, int N) {
  long t = (long)blockIdx.x * blockDim.x + threadIdx.x;
  if (t >= (long)N * HD_) return;
  int n = (int)(t >> 8), c = (int)(t & 255);
  pre[t] = agg[t] / nsum[n * HEADS_ + (c >> 5)] + bias[c];
}

// ---------------- BatchNorm (training mode, biased variance) ----------------
__global__ void bn_stats(const float* __restrict__ v, float* __restrict__ sums,
                         int nodesPerBlk, int N) {
  int c = threadIdx.x;                   // 256 channels
  int n0 = blockIdx.x * nodesPerBlk;
  float s = 0.f, s2 = 0.f;
  for (int i = 0; i < nodesPerBlk; ++i) {
    int n = n0 + i; if (n >= N) break;
    float x = v[(long)n * HD_ + c];
    s += x; s2 += x * x;
  }
  atomicAdd(&sums[c], s); atomicAdd(&sums[HD_ + c], s2);
}
__global__ void bn_finalize(const float* __restrict__ sums, float* __restrict__ stat, int N) {
  int c = threadIdx.x;
  float inv = 1.0f / (float)N;
  float mu = sums[c] * inv;
  float var = sums[HD_ + c] * inv - mu * mu;
  stat[c] = mu; stat[HD_ + c] = rsqrtf(var + EPS_);
}
__global__ void bn_apply(const float* __restrict__ pre, const float* __restrict__ stat,
                         const float* __restrict__ gamma, const float* __restrict__ beta,
                         float* __restrict__ h, int residual, int N) {
  long t = (long)blockIdx.x * blockDim.x + threadIdx.x;
  if (t >= (long)N * HD_) return;
  int c = (int)(t & 255);
  float x = (pre[t] - stat[c]) * stat[HD_ + c] * gamma[c] + beta[c];
  x = (x > 0.f) ? x : 0.f;               // ReLU
  h[t] = residual ? (x + h[t]) : x;
}

// ---------------- output heads ----------------
__global__ void head_out(const float* __restrict__ h, const float* __restrict__ mW,
                         const float* __restrict__ mb, const float* __restrict__ lW,
                         const float* __restrict__ lb, float* __restrict__ out, int N) {
  int n = blockIdx.x * blockDim.x + threadIdx.x;
  if (n >= N) return;
  const float* hr = h + (long)n * HD_;
  float a = 0.f, b = 0.f;
  for (int c = 0; c < HD_; ++c) { float v = hr[c]; a += v * mW[c]; b += v * lW[c]; }
  a += mb[0]; b += lb[0];
  b = fminf(10.0f, fmaxf(-10.0f, b));
  out[n] = a; out[N + n] = b;
}

// =====================================================================
extern "C" void kernel_launch(void* const* d_in, const int* in_sizes, int n_in,
                              void* d_out, int out_size, void* d_ws, size_t ws_size,
                              hipStream_t stream) {
  const float* x        = (const float*)d_in[0];
  const int*   ei       = (const int*)d_in[1];
  const float* in_W     = (const float*)d_in[2];
  const float* in_b     = (const float*)d_in[3];
  const float* in_gamma = (const float*)d_in[4];
  const float* in_beta  = (const float*)d_in[5];
  const float* Ws       = (const float*)d_in[6];
  const float* att_src  = (const float*)d_in[7];
  const float* att_dst  = (const float*)d_in[8];
  const float* conv_b   = (const float*)d_in[9];
  const float* bn_gamma = (const float*)d_in[10];
  const float* bn_beta  = (const float*)d_in[11];
  const float* mean_W   = (const float*)d_in[12];
  const float* mean_b   = (const float*)d_in[13];
  const float* lv_W     = (const float*)d_in[14];
  const float* lv_b     = (const float*)d_in[15];
  float* out = (float*)d_out;

  const int N = in_sizes[0] / INF_;      // 50000
  const int E = in_sizes[1] / 2;         // 800000
  const long NH = (long)N * HD_;
  const long ETOT = (long)E + N;

  // ---- bump allocator over d_ws (~220 MB total) ----
  char* ws = (char*)d_ws; size_t off = 0;
  auto alloc = [&](size_t bytes) -> void* {
    void* p = ws + off; off = (off + bytes + 255) & ~(size_t)255; return p;
  };
  float*          h     = (float*)alloc(NH * 4);
  float*          Z     = (float*)alloc(NH * 4);       // z / pre-BN (reused)
  float*          agg   = (float*)alloc(NH * 4);
  unsigned short* hb    = (unsigned short*)alloc(NH * 2);
  unsigned short* xb    = (unsigned short*)alloc((size_t)N * KP_ * 2);
  unsigned short* inWt  = (unsigned short*)alloc((size_t)HD_ * KP_ * 2);
  unsigned short* Wt    = (unsigned short*)alloc((size_t)LAYERS_ * HD_ * HD_ * 2);
  float*          es    = (float*)alloc((size_t)N * HEADS_ * 4);
  float*          ed    = (float*)alloc((size_t)N * HEADS_ * 4);
  unsigned*       nmax  = (unsigned*)alloc((size_t)N * HEADS_ * 4);
  float*          nsum  = (float*)alloc((size_t)N * HEADS_ * 4);
  float*          ee    = (float*)alloc((size_t)ETOT * HEADS_ * 4);
  float*          bnsum = (float*)alloc(2 * HD_ * 4);
  float*          bnst  = (float*)alloc(2 * HD_ * 4);
  (void)ws_size; (void)n_in; (void)out_size;

  const int BLK = 256;
  auto gb = [](long n, int b) { return (unsigned)((n + b - 1) / b); };

  // ---- weight staging (f32 -> bf16, transposed) ----
  cvt_x  <<<gb((long)N * KP_, BLK), BLK, 0, stream>>>(x, xb, N);
  cvt_wt <<<gb((long)HD_ * KP_, BLK), BLK, 0, stream>>>(in_W, inWt, INF_, HD_, KP_);
  for (int i = 0; i < LAYERS_; ++i)
    cvt_wt <<<gb((long)HD_ * HD_, BLK), BLK, 0, stream>>>(
        Ws + (size_t)i * HD_ * HD_, Wt + (size_t)i * HD_ * HD_, HD_, HD_, HD_);

  const long tiles = (long)(N / 16) * (HD_ / 16);      // 50000 (N % 16 == 0)
  const unsigned gemmBlocks = gb(tiles * 32, 128);
  const int nodesPerBlk = (N + 199) / 200;

  // ---- input projection: Linear -> BN -> ReLU ----
  gemm_bf16 <<<gemmBlocks, 128, 0, stream>>>(xb, inWt, in_b, Z, N, KP_, HD_);
  fill_f32 <<<gb(2 * HD_, BLK), BLK, 0, stream>>>(bnsum, 0.f, 2 * HD_);
  bn_stats <<<200, HD_, 0, stream>>>(Z, bnsum, nodesPerBlk, N);
  bn_finalize <<<1, HD_, 0, stream>>>(bnsum, bnst, N);
  bn_apply <<<gb(NH, BLK), BLK, 0, stream>>>(Z, bnst, in_gamma, in_beta, h, 0, N);

  // ---- GAT layers ----
  for (int i = 0; i < LAYERS_; ++i) {
    cvt_h <<<gb(NH, BLK), BLK, 0, stream>>>(h, hb, NH);
    gemm_bf16 <<<gemmBlocks, 128, 0, stream>>>(
        hb, Wt + (size_t)i * HD_ * HD_, nullptr, Z, N, HD_, HD_);
    attn_coef <<<gb((long)N * HEADS_, BLK), BLK, 0, stream>>>(
        Z, att_src + (size_t)i * HD_, att_dst + (size_t)i * HD_, es, ed, N);
    fill_f32 <<<gb(NH, BLK), BLK, 0, stream>>>(agg, 0.f, NH);
    fill_f32 <<<gb((long)N * HEADS_, BLK), BLK, 0, stream>>>(nsum, 0.f, (long)N * HEADS_);
    fill_u32 <<<gb((long)N * HEADS_, BLK), BLK, 0, stream>>>(nmax, ENC_NEG_INF, (long)N * HEADS_);
    edge_logits <<<gb(ETOT * HEADS_, BLK), BLK, 0, stream>>>(ei, es, ed, ee, nmax, E, N);
    edge_scatter <<<gb(ETOT * 32, BLK), BLK, 0, stream>>>(ei, ee, nmax, Z, nsum, agg, E, N);
    finalize_agg <<<gb(NH, BLK), BLK, 0, stream>>>(agg, nsum, conv_b + (size_t)i * HD_, Z, N);
    fill_f32 <<<gb(2 * HD_, BLK), BLK, 0, stream>>>(bnsum, 0.f, 2 * HD_);
    bn_stats <<<200, HD_, 0, stream>>>(Z, bnsum, nodesPerBlk, N);
    bn_finalize <<<1, HD_, 0, stream>>>(bnsum, bnst, N);
    bn_apply <<<gb(NH, BLK), BLK, 0, stream>>>(
        Z, bnst, bn_gamma + (size_t)i * HD_, bn_beta + (size_t)i * HD_, h, 1, N);
  }

  // ---- output heads ----
  head_out <<<gb(N, BLK), BLK, 0, stream>>>(h, mean_W, mean_b, lv_W, lv_b, out, N);
}
